// SetPooling_72164040507939
// MI455X (gfx1250) — matrix operand order repaired and checked by reading, not compiled, not run on hardware
//
#include <hip/hip_runtime.h>

// ---------------------------------------------------------------------------
// SetPooling (SetTransformer encoder/decoder) for gfx1250 (MI455X).
// Heavy GEMMs: bf16 WMMA (v_wmma_f32_16x16x32_bf16), fp32 accumulation.
// Block tile 128x64x32, 8 waves, 32x32 per wave (4 WMMA / wave / K-step).
// Fast path (full tiles): vectorized b128 global loads + b128 LDS stores,
// no divergence. Guarded path (edge tiles): clamped addresses + cndmask.
// ---------------------------------------------------------------------------

#define B_    64
#define N_    1024
#define D_    512
#define H_    4
#define M_    32
#define DH_   128
#define DFF_  512
#define SCALE_ 0.08838834764831845f   // 1/sqrt(128)

#define BM_T 128
#define BN_T 64
#define BK_T 32
#define LDSTR 40     // LDS row stride in bf16 elements (80B: 16B-aligned, padded)

typedef __bf16 bf16;
typedef __attribute__((ext_vector_type(16))) __bf16 v16bf;
typedef __attribute__((ext_vector_type(8)))  __bf16 v8bf;
typedef __attribute__((ext_vector_type(4)))  __bf16 v4bf;
typedef __attribute__((ext_vector_type(8)))  float  v8f;
typedef __attribute__((ext_vector_type(4)))  float  v4f;

union FragU { v16bf v; v8bf h[2]; };
union V8U  { v8bf  v; v4bf h[2]; };

__device__ __forceinline__ v8bf pack8(v4f a, v4f b) {
    V8U u;
    u.h[0] = __builtin_convertvector(a, v4bf);
    u.h[1] = __builtin_convertvector(b, v4bf);
    return u.v;
}

// 32x32 wave tile: 2 A-frags x 2 B-frags -> 4 WMMA
__device__ __forceinline__ void wave_mma(const bf16* Asm, const bf16* Bsm,
                                         int rw, int cw, int lane,
                                         v8f& a00, v8f& a01, v8f& a10, v8f& a11)
{
    const int half = lane >> 4;
    const int lc   = lane & 15;
    FragU fa0, fa1, fb0, fb1;
    const v8bf* ar0 = (const v8bf*)(Asm + (rw + lc) * LDSTR);
    const v8bf* ar1 = (const v8bf*)(Asm + (rw + 16 + lc) * LDSTR);
    // A layout: VGPR0..3 <- K = half*8+{0..7}; VGPR4..7 <- K = 16+half*8+{0..7}
    fa0.h[0] = ar0[half];     fa0.h[1] = ar0[2 + half];
    fa1.h[0] = ar1[half];     fa1.h[1] = ar1[2 + half];
    const v8bf* br0 = (const v8bf*)(Bsm + (cw + lc) * LDSTR);
    const v8bf* br1 = (const v8bf*)(Bsm + (cw + 16 + lc) * LDSTR);
    // B layout: VGPR0..3 <- K = half*8+{0..7}; VGPR4..7 <- K = 16+half*8+{0..7}
    fb0.h[0] = br0[half]; fb0.h[1] = br0[2 + half];
    fb1.h[0] = br1[half]; fb1.h[1] = br1[2 + half];
    a00 = __builtin_amdgcn_wmma_f32_16x16x32_bf16(false, fa0.v, false, fb0.v, (short)0, a00, false, false);
    a01 = __builtin_amdgcn_wmma_f32_16x16x32_bf16(false, fa0.v, false, fb1.v, (short)0, a01, false, false);
    a10 = __builtin_amdgcn_wmma_f32_16x16x32_bf16(false, fa1.v, false, fb0.v, (short)0, a10, false, false);
    a11 = __builtin_amdgcn_wmma_f32_16x16x32_bf16(false, fa1.v, false, fb1.v, (short)0, a11, false, false);
}

__device__ __forceinline__ void store_tile(float* Cb, v8f acc, int rbase, int col,
                                           long sCm, int Mrows, int Ncols,
                                           const float* bias, float alpha, int relu,
                                           bool guard)
{
    if (guard && col >= Ncols) return;
    const float bX = bias ? bias[col] : 0.f;
#pragma unroll
    for (int i = 0; i < 8; ++i) {
        const int rr = rbase + i;
        if (guard && rr >= Mrows) continue;
        float v = acc[i] * alpha + bX;
        if (relu) v = fmaxf(v, 0.f);
        Cb[(long)rr * sCm + col] = v;
    }
}

// ---------------------------------------------------------------------------
// General batched GEMM:  C[z](m,n) = act( alpha * sum_k A[z](m,k)*B[z](k,n) + bias[n] )
//   z = (b,h) via Hh; A k-contiguous; B has sBk==1 (k-contig) or sBn==1.
// ---------------------------------------------------------------------------
__global__ __launch_bounds__(256)
void gemm_bf16_wmma(const float* __restrict__ A, const float* __restrict__ Bm,
                    const float* __restrict__ bias, float* __restrict__ C,
                    int Mrows, int Ncols, int Kdim,
                    long sAb, long sAh, long sAm,
                    long sBb, long sBh, long sBk, long sBn,
                    long sCb, long sCh, long sCm,
                    int Hh, float alpha, int relu)
{
    __shared__ __align__(16) bf16 Asm[BM_T * LDSTR];   // [m][k]
    __shared__ __align__(16) bf16 Bsm[BN_T * LDSTR];   // [n][k] (transposed staging)

    const int z  = blockIdx.z;
    const int bb = z / Hh;
    const int hh = z - bb * Hh;
    const float* Ab = A  + (long)bb * sAb + (long)hh * sAh;
    const float* Bb = Bm + (long)bb * sBb + (long)hh * sBh;
    float*       Cb = C  + (long)bb * sCb + (long)hh * sCh;

    const int m0   = blockIdx.y * BM_T;
    const int n0   = blockIdx.x * BN_T;
    const int tid  = threadIdx.x;
    const int wave = tid >> 5;
    const int lane = tid & 31;
    const int rw   = (wave >> 1) * 32;    // wave row offset in tile (0,32,64,96)
    const int cw   = (wave & 1) * 32;     // wave col offset in tile (0,32)
    const int half = lane >> 4;
    const int lc   = lane & 15;

    v8f a00 = {0.f,0.f,0.f,0.f,0.f,0.f,0.f,0.f};
    v8f a01 = a00, a10 = a00, a11 = a00;

    const int ksteps = (Kdim + 31) >> 5;
    const bool fast = (m0 + BM_T <= Mrows) && (n0 + BN_T <= Ncols) && ((Kdim & 31) == 0);

    if (fast) {
        // ------------------------- fast path (no guards) -------------------
        const int arow0 = tid >> 2;             // A slot 0 row (0..63)
        const int akk   = (tid & 3) * 8;        // A k chunk
        for (int ks = 0; ks < ksteps; ++ks) {
            const int k0 = ks * 32;
            // A tile: 128x32, two slots per thread, b128-pair loads
#pragma unroll
            for (int it = 0; it < 2; ++it) {
                const int row = arow0 + 64 * it;
                const v4f* ap = (const v4f*)(Ab + (long)(m0 + row) * sAm + (k0 + akk));
                if (ks + 1 < ksteps) __builtin_prefetch((const float*)ap + 32, 0, 1);
                *(v8bf*)(Asm + row * LDSTR + akk) = pack8(ap[0], ap[1]);
            }
            // B tile: 32x64 -> Bsm[n][k]
            if (sBk == 1) {            // k-contiguous rows (attention scores)
                const int nrow = tid >> 2;
                const int kk   = (tid & 3) * 8;
                const v4f* bp = (const v4f*)(Bb + (long)(n0 + nrow) * sBn + (k0 + kk));
                if (ks + 1 < ksteps) __builtin_prefetch((const float*)bp + 32, 0, 1);
                *(v8bf*)(Bsm + nrow * LDSTR + kk) = pack8(bp[0], bp[1]);
            } else {                   // n-contiguous rows (weights / V): transpose into LDS
                const int kr = tid >> 3;            // 0..31
                const int nn = (tid & 7) * 8;       // 0..56
                const v4f* bp = (const v4f*)(Bb + (long)(k0 + kr) * sBk + (n0 + nn));
                if (ks + 1 < ksteps) __builtin_prefetch((const float*)bp + 32 * sBk, 0, 1);
                const v4f b0 = bp[0], b1 = bp[1];
#pragma unroll
                for (int j = 0; j < 4; ++j) {
                    Bsm[(nn + j)     * LDSTR + kr] = (bf16)b0[j];
                    Bsm[(nn + 4 + j) * LDSTR + kr] = (bf16)b1[j];
                }
            }
            __syncthreads();
            wave_mma(Asm, Bsm, rw, cw, lane, a00, a01, a10, a11);
            __syncthreads();
        }
        const int r0 = m0 + rw + half * 8;
        const int c0 = n0 + cw + lc;
        store_tile(Cb, a00, r0,      c0,      sCm, Mrows, Ncols, bias, alpha, relu, false);
        store_tile(Cb, a01, r0,      c0 + 16, sCm, Mrows, Ncols, bias, alpha, relu, false);
        store_tile(Cb, a10, r0 + 16, c0,      sCm, Mrows, Ncols, bias, alpha, relu, false);
        store_tile(Cb, a11, r0 + 16, c0 + 16, sCm, Mrows, Ncols, bias, alpha, relu, false);
    } else {
        // --------------------- guarded path (edge tiles) -------------------
        // Branch-free: clamp addresses into range, select 0 via cndmask.
        const int arow0 = tid >> 2;
        const int akk   = (tid & 3) * 8;
        const int brow  = tid >> 2;             // B: n row, strided-k scalar loads
        const int bkk   = (tid & 3) * 8;
        for (int ks = 0; ks < ksteps; ++ks) {
            const int k0 = ks * 32;
#pragma unroll
            for (int it = 0; it < 2; ++it) {
                const int row  = arow0 + 64 * it;
                const bool rOK = (m0 + row) < Mrows;
                const int rcl  = rOK ? (m0 + row) : (Mrows - 1);
                const float* aptr = Ab + (long)rcl * sAm;
                bf16* dst = Asm + row * LDSTR + akk;
#pragma unroll
                for (int j = 0; j < 8; ++j) {
                    const int kg   = k0 + akk + j;
                    const bool kOK = kg < Kdim;
                    const float vv = aptr[kOK ? kg : (Kdim - 1)];
                    dst[j] = (bf16)((rOK && kOK) ? vv : 0.0f);
                }
            }
            {
                const bool nOK = (n0 + brow) < Ncols;
                const int ncl  = nOK ? (n0 + brow) : (Ncols - 1);
                const float* bptr = Bb + (long)ncl * sBn;
                bf16* dst = Bsm + brow * LDSTR + bkk;
#pragma unroll
                for (int j = 0; j < 8; ++j) {
                    const int kg   = k0 + bkk + j;
                    const bool kOK = kg < Kdim;
                    const float vv = bptr[(long)(kOK ? kg : (Kdim - 1)) * sBk];
                    dst[j] = (bf16)((nOK && kOK) ? vv : 0.0f);
                }
            }
            __syncthreads();
            wave_mma(Asm, Bsm, rw, cw, lane, a00, a01, a10, a11);
            __syncthreads();
        }
        const int r0 = m0 + rw + half * 8;
        const int c0 = n0 + cw + lc;
        store_tile(Cb, a00, r0,      c0,      sCm, Mrows, Ncols, bias, alpha, relu, true);
        store_tile(Cb, a01, r0,      c0 + 16, sCm, Mrows, Ncols, bias, alpha, relu, true);
        store_tile(Cb, a10, r0 + 16, c0,      sCm, Mrows, Ncols, bias, alpha, relu, true);
        store_tile(Cb, a11, r0 + 16, c0 + 16, sCm, Mrows, Ncols, bias, alpha, relu, true);
    }
}

// ---------------------------------------------------------------------------
// Row softmax (in place), one wave32 per row.
// ---------------------------------------------------------------------------
__global__ __launch_bounds__(256)
void softmax_rows(float* __restrict__ E, int nrows, int len)
{
    const int row  = blockIdx.x * 8 + (threadIdx.x >> 5);
    const int lane = threadIdx.x & 31;
    if (row >= nrows) return;
    float* p = E + (long)row * len;

    float mx = -3.0e38f;
    for (int i = lane; i < len; i += 32) mx = fmaxf(mx, p[i]);
    for (int o = 16; o; o >>= 1) mx = fmaxf(mx, __shfl_xor(mx, o, 32));
    float s = 0.f;
    for (int i = lane; i < len; i += 32) { float e = __expf(p[i] - mx); p[i] = e; s += e; }
    for (int o = 16; o; o >>= 1) s += __shfl_xor(s, o, 32);
    const float inv = 1.0f / s;
    for (int i = lane; i < len; i += 32) p[i] *= inv;
}

// ---------------------------------------------------------------------------
// Fused residual + LayerNorm over D=512:  Y[row] = LN(X[row] + R[rrow])*g + b
//   resMod>0: residual row = row % resMod (broadcast inducing points / seed).
// ---------------------------------------------------------------------------
__global__ __launch_bounds__(256)
void ln_residual(const float* __restrict__ X, const float* __restrict__ R,
                 const float* __restrict__ g, const float* __restrict__ b,
                 float* __restrict__ Y, int nrows, int resMod)
{
    const int row  = blockIdx.x * 8 + (threadIdx.x >> 5);
    const int lane = threadIdx.x & 31;
    if (row >= nrows) return;
    const float* x = X + (long)row * D_;
    const int rrow = resMod ? (row % resMod) : row;
    const float* r = R + (long)rrow * D_;

    float v[16];
    float s = 0.f;
#pragma unroll
    for (int i = 0; i < 16; ++i) { v[i] = x[lane + 32 * i] + r[lane + 32 * i]; s += v[i]; }
    for (int o = 16; o; o >>= 1) s += __shfl_xor(s, o, 32);
    const float mu = s * (1.0f / D_);
    float q = 0.f;
#pragma unroll
    for (int i = 0; i < 16; ++i) { const float d = v[i] - mu; q += d * d; }
    for (int o = 16; o; o >>= 1) q += __shfl_xor(q, o, 32);
    const float rstd = rsqrtf(q * (1.0f / D_) + 1e-5f);
    float* y = Y + (long)row * D_;
#pragma unroll
    for (int i = 0; i < 16; ++i) {
        const int c = lane + 32 * i;
        y[c] = (v[i] - mu) * rstd * g[c] + b[c];
    }
}

// ---------------------------------------------------------------------------
// Host-side helpers (enqueue on stream only; graph-capture safe)
// ---------------------------------------------------------------------------
static inline void launch_gemm(hipStream_t s,
    const float* A, const float* Bm, const float* bias, float* C,
    int Mr, int Nc, int Kd,
    long sAb, long sAh, long sAm,
    long sBb, long sBh, long sBk, long sBn,
    long sCb, long sCh, long sCm,
    int Hh, int nb, float alpha, int relu)
{
    dim3 grid((Nc + BN_T - 1) / BN_T, (Mr + BM_T - 1) / BM_T, nb);
    gemm_bf16_wmma<<<grid, dim3(256), 0, s>>>(A, Bm, bias, C, Mr, Nc, Kd,
        sAb, sAh, sAm, sBb, sBh, sBk, sBn, sCb, sCh, sCm, Hh, alpha, relu);
}

static inline void launch_softmax(hipStream_t s, float* E, int nrows, int len)
{
    softmax_rows<<<dim3((nrows + 7) / 8), dim3(256), 0, s>>>(E, nrows, len);
}

static inline void launch_ln(hipStream_t s, const float* X, const float* R,
                             const float* g, const float* b, float* Y,
                             int nrows, int resMod)
{
    ln_residual<<<dim3((nrows + 7) / 8), dim3(256), 0, s>>>(X, R, g, b, Y, nrows, resMod);
}

// DGL MultiHeadAttention block (MAB): LN(x+attn) then LN(h+FFN(h)).
// x: broadcast ? (Lx,D) shared over batch : (B*Lx,D).  mem: (B*Lm,D).
// param base pb: Wq,Wk,Wv,Wo,W1,b1,W2,b2,ln1_g,ln1_b,ln2_g,ln2_b
static void run_mha(hipStream_t s,
                    const float* x, int Lx, int xB, const float* mem, int Lm,
                    void* const* din, int pb,
                    float* qb, float* kb, float* vb, float* eb, float* oh,
                    float* o2, float* h1, float* f1, float* f2, float* outp)
{
    const float* Wq  = (const float*)din[pb + 0];
    const float* Wk  = (const float*)din[pb + 1];
    const float* Wv  = (const float*)din[pb + 2];
    const float* Wo  = (const float*)din[pb + 3];
    const float* W1  = (const float*)din[pb + 4];
    const float* b1  = (const float*)din[pb + 5];
    const float* W2  = (const float*)din[pb + 6];
    const float* b2  = (const float*)din[pb + 7];
    const float* l1g = (const float*)din[pb + 8];
    const float* l1b = (const float*)din[pb + 9];
    const float* l2g = (const float*)din[pb + 10];
    const float* l2b = (const float*)din[pb + 11];

    const int BLx = B_ * Lx;
    const int qR  = xB ? Lx : BLx;

    // projections (plain GEMMs, weights row-major KxN)
    launch_gemm(s, x,   Wq, nullptr, qb, qR,      D_, D_, 0,0,D_, 0,0,(long)D_,1, 0,0,D_, 1,1, 1.f, 0);
    launch_gemm(s, mem, Wk, nullptr, kb, B_ * Lm, D_, D_, 0,0,D_, 0,0,(long)D_,1, 0,0,D_, 1,1, 1.f, 0);
    launch_gemm(s, mem, Wv, nullptr, vb, B_ * Lm, D_, D_, 0,0,D_, 0,0,(long)D_,1, 0,0,D_, 1,1, 1.f, 0);
    // scores: e[b,h] = (q_bh) (k_bh)^T * SCALE        (B k-contiguous: sBk=1)
    launch_gemm(s, qb, kb, nullptr, eb, Lx, Lm, DH_,
                xB ? 0L : (long)Lx * D_, DH_, D_,
                (long)Lm * D_, DH_, 1, D_,
                (long)H_ * Lx * Lm, (long)Lx * Lm, Lm,
                H_, B_ * H_, SCALE_, 0);
    launch_softmax(s, eb, B_ * H_ * Lx, Lm);
    // o heads: (Lx x Lm)@(Lm x DH) -> oh laid out (b, m, h, dh)
    launch_gemm(s, eb, vb, nullptr, oh, Lx, DH_, Lm,
                (long)H_ * Lx * Lm, (long)Lx * Lm, Lm,
                (long)Lm * D_, DH_, D_, 1,
                (long)Lx * D_, DH_, D_,
                H_, B_ * H_, 1.f, 0);
    // output projection + residual LN (intra)
    launch_gemm(s, oh, Wo, nullptr, o2, BLx, D_, D_, 0,0,D_, 0,0,(long)D_,1, 0,0,D_, 1,1, 1.f, 0);
    launch_ln(s, o2, x, l1g, l1b, h1, BLx, xB ? Lx : 0);
    // FFN + residual LN (inter)
    launch_gemm(s, h1, W1, b1, f1, BLx, DFF_, D_,   0,0,D_,   0,0,(long)DFF_,1, 0,0,DFF_, 1,1, 1.f, 1);
    launch_gemm(s, f1, W2, b2, f2, BLx, D_,   DFF_, 0,0,DFF_, 0,0,(long)D_,  1, 0,0,D_,   1,1, 1.f, 0);
    launch_ln(s, f2, h1, l2g, l2b, outp, BLx, 0);
}

// ---------------------------------------------------------------------------
// Input index map (setup_inputs dict order, flattened):
//  0 feats | 1 isab0.ind | 2-13 isab0.mha0 | 14-25 isab0.mha1
//  26 isab1.ind | 27-38 isab1.mha0 | 39-50 isab1.mha1
//  51 pma.seed | 52-55 pma.ffn (W1,b1,W2,b2) | 56-67 pma.mha | 68-79 sab
// ---------------------------------------------------------------------------
extern "C" void kernel_launch(void* const* d_in, const int* in_sizes, int n_in,
                              void* d_out, int out_size, void* d_ws, size_t ws_size,
                              hipStream_t stream)
{
    (void)in_sizes; (void)n_in; (void)out_size; (void)ws_size;
    const float* feats = (const float*)d_in[0];
    float* ws = (float*)d_ws;

    const long BND = (long)B_ * N_ * D_;            // 33,554,432 floats
    const long BMD = (long)B_ * M_ * D_;            //  1,048,576 floats
    const long ESZ = (long)B_ * H_ * N_ * M_;       //  8,388,608 floats
    float* X0 = ws;            // current h
    float* X1 = X0 + BND;      // next h / hf / temps
    float* KB = X1 + BND;      // k proj (large)
    float* VB = KB + BND;      // v proj (large)
    float* QB = VB + BND;      // q / o_heads / ffn hidden (large)
    float* EB = QB + BND;      // attention scores
    float* S1 = EB + ESZ;      // small (B*M*D)
    float* S2 = S1 + BMD;
    float* S3 = S2 + BMD;

    const float* hcur = feats;
    for (int l = 0; l < 2; ++l) {
        const float* ind = (const float*)d_in[l == 0 ? 1 : 26];
        const int p0 = (l == 0) ? 2 : 27;   // mha0: MAB(I, X)
        const int p1 = (l == 0) ? 14 : 39;  // mha1: MAB(X, mem)
        // inducing points attend to nodes -> mem (B,M,D) in S2
        run_mha(stream, ind, M_, 1, hcur, N_, d_in, p0,
                S1, KB, VB, EB, S2, S3, S1, S2, S3, S2);
        // nodes attend to induced summary -> new h (B,N,D) in X0
        run_mha(stream, hcur, N_, 0, S2, M_, d_in, p1,
                KB, S1, S3, EB, QB, VB, X1, QB, VB, X0);
        hcur = X0;
    }

    // --- Decoder: PMA = MAB(seed, rFF(h)) ---
    const float* W1f = (const float*)d_in[52];
    const float* b1f = (const float*)d_in[53];
    const float* W2f = (const float*)d_in[54];
    const float* b2f = (const float*)d_in[55];
    launch_gemm(stream, X0, W1f, b1f, QB, B_ * N_, DFF_, D_,
                0,0,D_, 0,0,(long)DFF_,1, 0,0,DFF_, 1,1, 1.f, 1);
    launch_gemm(stream, QB, W2f, b2f, X1, B_ * N_, D_, DFF_,
                0,0,DFF_, 0,0,(long)D_,1, 0,0,D_, 1,1, 1.f, 0);   // hf in X1
    const float* seed = (const float*)d_in[51];
    run_mha(stream, seed, 1, 1, X1, N_, d_in, 56,
            S1, KB, VB, EB, S2, S3, S1, S2, S3, S2);              // pooled (B,1,D) in S2

    // --- SAB over the pooled k=1 vectors; final LN writes d_out (64 x 512) ---
    run_mha(stream, S2, 1, 0, S2, 1, d_in, 68,
            S1, KB, VB, EB, QB, X1, S3, QB, X1, (float*)d_out);
}
